// TimeMixing_39152921870688
// MI455X (gfx1250) — matrix operand order repaired and checked by reading, not compiled, not run on hardware
//
#include <hip/hip_runtime.h>

// Problem constants (match reference)
#define BB 8
#define TT 2048
#define CC 2048
constexpr int Mdim = BB * TT;   // 16384 rows
constexpr int Ndim = CC;        // 2048
constexpr int Kdim = CC;        // 2048

// WMMA vector types (CDNA5, wave32)
typedef __attribute__((ext_vector_type(16))) __bf16 bf16x16;
typedef __attribute__((ext_vector_type(8)))  float  f32x8;
typedef __attribute__((ext_vector_type(4)))  int    v4i;

#define LDS_AS __attribute__((address_space(3)))
#define GLB_AS __attribute__((address_space(1)))

#if __has_builtin(__builtin_amdgcn_global_load_async_to_lds_b128) && \
    __has_builtin(__builtin_amdgcn_s_wait_asynccnt)
#define HAVE_ASYNC_LDS 1
#else
#define HAVE_ASYNC_LDS 0
#endif

__device__ __forceinline__ void copy16_g2l(const unsigned short* g, unsigned short* l) {
#if HAVE_ASYNC_LDS
    __builtin_amdgcn_global_load_async_to_lds_b128(
        (GLB_AS v4i*)g, (LDS_AS v4i*)l, 0, 0);
#else
    *(uint4*)l = *(const uint4*)g;
#endif
}

__device__ __forceinline__ void wait_async_done() {
#if HAVE_ASYNC_LDS
    __builtin_amdgcn_s_wait_asynccnt(0);
#endif
}

__device__ __forceinline__ unsigned short f2bf(float f) {
    unsigned int u = __float_as_uint(f);
    u += 0x7FFFu + ((u >> 16) & 1u);   // round-to-nearest-even
    return (unsigned short)(u >> 16);
}

// ---------------------------------------------------------------------------
// f32 -> bf16 cast (vectorized 4-wide)
// ---------------------------------------------------------------------------
__global__ __launch_bounds__(256)
void cast_f32_bf16(const float* __restrict__ in, unsigned short* __restrict__ out, int n) {
    int i = (blockIdx.x * blockDim.x + threadIdx.x) * 4;
    if (i + 3 < n) {
        float4 f = *(const float4*)(in + i);
        ushort4 o;
        o.x = f2bf(f.x); o.y = f2bf(f.y); o.z = f2bf(f.z); o.w = f2bf(f.w);
        *(ushort4*)(out + i) = o;
    } else {
        for (; i < n; ++i) out[i] = f2bf(in[i]);
    }
}

// ---------------------------------------------------------------------------
// bf16 WMMA GEMM, NT:  C[M,N] = A[M,K] * Bw[N,K]^T   (f32 accumulate)
// WG tile: 256(M) x 128(N), TK=32 K-slab, double-buffered async LDS staging.
// 8 waves as 4x2; each wave owns 64x64 = 4x4 WMMA tiles (16 wmma / K-step,
// 8 fragment loads / K-step).
// ---------------------------------------------------------------------------
#define TM 256
#define TN 128
#define TK 32
#define LDP 40   // padded LDS row stride (bf16 elems): 80B rows break bank conflicts

__global__ __launch_bounds__(256)
void gemm_bf16_nt(const unsigned short* __restrict__ A,
                  const unsigned short* __restrict__ Bw,
                  float* __restrict__ C,
                  int M, int N, int K) {
    __shared__ __align__(16) unsigned short lA[2][TM][LDP];   // 2*256*40*2 = 40960 B
    __shared__ __align__(16) unsigned short lB[2][TN][LDP];   // 2*128*40*2 = 20480 B

    const int tid  = threadIdx.x;
    const int wid  = tid >> 5;
    const int lane = tid & 31;

    const int m0 = blockIdx.y * TM;
    const int n0 = blockIdx.x * TN;

    const int wm = (wid >> 1) * 64;    // wave M origin in tile: 0/64/128/192
    const int wn = (wid & 1) * 64;     // wave N origin in tile: 0/64

    const int half = lane >> 4;        // lane half selects K sub-blocks
    const int mrow = lane & 15;        // A row within 16x32 fragment

    f32x8 acc[4][4];
    #pragma unroll
    for (int i = 0; i < 4; ++i)
        #pragma unroll
        for (int j = 0; j < 4; ++j)
            #pragma unroll
            for (int e = 0; e < 8; ++e) acc[i][j][e] = 0.0f;

    // staging map: 8-elem (16B) chunks. A: 256 rows x 4 chunks = 1024; B: 512.
    const int ar[4] = { (tid          ) >> 2, (tid + 256) >> 2,
                        (tid + 512    ) >> 2, (tid + 768) >> 2 };
    const int ac    = (tid & 3) * 8;
    const int br[2] = { (tid) >> 2, (tid + 256) >> 2 };

    const int nt = K / TK;

    // prologue: stage tile 0 into buffer 0
    {
        #pragma unroll
        for (int t = 0; t < 4; ++t)
            copy16_g2l(A + (size_t)(m0 + ar[t]) * K + ac, &lA[0][ar[t]][ac]);
        #pragma unroll
        for (int t = 0; t < 2; ++t)
            copy16_g2l(Bw + (size_t)(n0 + br[t]) * K + ac, &lB[0][br[t]][ac]);
    }

    for (int kt = 0; kt < nt; ++kt) {
        const int cur = kt & 1;
        wait_async_done();        // this wave's tile-kt DMA landed
        __syncthreads();          // all waves' DMA landed; all compute on buf cur^1 done

        if (kt + 1 < nt) {        // overlap tile kt+1 DMA with tile kt compute
            const int nxt = cur ^ 1;
            const int kofs = (kt + 1) * TK;
            #pragma unroll
            for (int t = 0; t < 4; ++t)
                copy16_g2l(A + (size_t)(m0 + ar[t]) * K + kofs + ac, &lA[nxt][ar[t]][ac]);
            #pragma unroll
            for (int t = 0; t < 2; ++t)
                copy16_g2l(Bw + (size_t)(n0 + br[t]) * K + kofs + ac, &lB[nxt][br[t]][ac]);
        }

        // A fragments (16x32): lane<16 -> K 0-7 & 16-23 ; lane>=16 -> K 8-15 & 24-31
        union { bf16x16 v; uint4 q[2]; } fa[4];
        #pragma unroll
        for (int i = 0; i < 4; ++i) {
            const unsigned short* p = &lA[cur][wm + i * 16 + mrow][half * 8];
            fa[i].q[0] = *(const uint4*)(p);
            fa[i].q[1] = *(const uint4*)(p + 16);
        }
        // B fragments (32x16): lane -> col N=lane&15; lane<16 -> K 0-7 & 16-23 ; lane>=16 -> K 8-15 & 24-31
        union { bf16x16 v; uint4 q[2]; } fb[4];
        #pragma unroll
        for (int j = 0; j < 4; ++j) {
            const unsigned short* p = &lB[cur][wn + j * 16 + (lane & 15)][half * 8];
            fb[j].q[0] = *(const uint4*)(p);
            fb[j].q[1] = *(const uint4*)(p + 16);
        }

        #pragma unroll
        for (int i = 0; i < 4; ++i)
            #pragma unroll
            for (int j = 0; j < 4; ++j)
                acc[i][j] = __builtin_amdgcn_wmma_f32_16x16x32_bf16(
                    false, fa[i].v, false, fb[j].v, (short)0, acc[i][j], false, false);
    }

    // Store D: VGPR e, lane -> M = (lane>=16?8:0)+e, N = lane&15
    const int nn = lane & 15;
    const int mh = half * 8;
    #pragma unroll
    for (int i = 0; i < 4; ++i)
        #pragma unroll
        for (int j = 0; j < 4; ++j) {
            float* cp = C + (size_t)(m0 + wm + i * 16 + mh) * N + (n0 + wn + j * 16 + nn);
            #pragma unroll
            for (int e = 0; e < 8; ++e) cp[(size_t)e * N] = acc[i][j][e];
        }
}

// ---------------------------------------------------------------------------
// WKV sequential scan: one lane per (b,c) recurrence; coalesced in c.
// Emits bf16 directly for the Wo GEMM.
// ---------------------------------------------------------------------------
__global__ __launch_bounds__(256)
void wkv_scan(const float* __restrict__ r, const float* __restrict__ k,
              const float* __restrict__ v,
              const float* __restrict__ time_decay, const float* __restrict__ time_first,
              unsigned short* __restrict__ outbf) {
    const int idx = blockIdx.x * blockDim.x + threadIdx.x;   // b*C + c
    const int c = idx % CC;
    const int b = idx / CC;
    if (b >= BB) return;

    const float w  = __expf(-__expf(time_decay[c]));
    const float tf = __expf(time_first[c]);

    float a = 0.0f, bb = 0.0f;
    size_t off = (size_t)b * TT * CC + c;

    #pragma unroll 4
    for (int t = 0; t < TT; ++t, off += CC) {
        const float rt = r[off];
        const float kt = k[off];
        const float vt = v[off];
        const float ek  = __expf(kt);
        const float eww = __expf(tf * kt);
        const float sr  = 1.0f / (1.0f + __expf(-rt));
        const float o   = sr * (a + eww * vt) / (bb + eww + 1e-6f);
        a  = w * a  + ek * vt;
        bb = w * bb + ek;
        outbf[off] = f2bf(o);
    }
}

// ---------------------------------------------------------------------------
extern "C" void kernel_launch(void* const* d_in, const int* in_sizes, int n_in,
                              void* d_out, int out_size, void* d_ws, size_t ws_size,
                              hipStream_t stream) {
    const float* x   = (const float*)d_in[0];
    const float* Wr  = (const float*)d_in[1];
    const float* Wk  = (const float*)d_in[2];
    const float* Wv  = (const float*)d_in[3];
    const float* Wo  = (const float*)d_in[4];
    const float* td  = (const float*)d_in[5];
    const float* tfp = (const float*)d_in[6];

    const size_t nX = (size_t)Mdim * Kdim;    // 33,554,432
    const size_t nW = (size_t)Ndim * Kdim;    //  4,194,304

    // workspace carve-out
    char* p = (char*)d_ws;
    unsigned short* xb  = (unsigned short*)p; p += nX * 2;
    unsigned short* wrb = (unsigned short*)p; p += nW * 2;
    unsigned short* wkb = (unsigned short*)p; p += nW * 2;
    unsigned short* wvb = (unsigned short*)p; p += nW * 2;
    unsigned short* wob = (unsigned short*)p; p += nW * 2;
    float* rr = (float*)p; p += nX * 4;
    float* kk = (float*)p; p += nX * 4;
    float* vv = (float*)p; p += nX * 4;
    unsigned short* ob = (unsigned short*)p; p += nX * 2;

    // 1) casts to bf16
    cast_f32_bf16<<<(int)(nX / 4 / 256), 256, 0, stream>>>(x,  xb,  (int)nX);
    cast_f32_bf16<<<(int)(nW / 4 / 256), 256, 0, stream>>>(Wr, wrb, (int)nW);
    cast_f32_bf16<<<(int)(nW / 4 / 256), 256, 0, stream>>>(Wk, wkb, (int)nW);
    cast_f32_bf16<<<(int)(nW / 4 / 256), 256, 0, stream>>>(Wv, wvb, (int)nW);
    cast_f32_bf16<<<(int)(nW / 4 / 256), 256, 0, stream>>>(Wo, wob, (int)nW);

    // 2) r/k/v projections (WMMA bf16, f32 accumulate)
    dim3 ggrid(Ndim / TN, Mdim / TM);   // 16 x 64
    gemm_bf16_nt<<<ggrid, 256, 0, stream>>>(xb, wrb, rr, Mdim, Ndim, Kdim);
    gemm_bf16_nt<<<ggrid, 256, 0, stream>>>(xb, wkb, kk, Mdim, Ndim, Kdim);
    gemm_bf16_nt<<<ggrid, 256, 0, stream>>>(xb, wvb, vv, Mdim, Ndim, Kdim);

    // 3) WKV scan
    wkv_scan<<<(BB * CC) / 256, 256, 0, stream>>>(rr, kk, vv, td, tfp, ob);

    // 4) output projection -> d_out (f32)
    gemm_bf16_nt<<<ggrid, 256, 0, stream>>>(ob, wob, (float*)d_out, Mdim, Ndim, Kdim);
}
